// Generative_55559696941094
// MI455X (gfx1250) — matrix operand order repaired and checked by reading, not compiled, not run on hardware
//
#include <hip/hip_runtime.h>
#include <hip/hip_bf16.h>

// ---------------------------------------------------------------------------
// SampleRNN 3-tier generative module for MI455X (gfx1250, wave32).
//
// Roofline: ~3.75 GB of fp32 weights streamed ~once, ~3.2 GFLOP total ->
// memory bound (~1 FLOP/byte).  Tier-3 MLPs (3.2 GB) and the hoisted tier-2
// input-gate GEMM (134 MB) are done with v_wmma_f32_16x16x32_f16 (f32 accum,
// f16 convert in-register; weight bandwidth unchanged).  The sequential LSTM
// chains are latency-bound persistent single-workgroup kernels with LDS state.
//
// v3: B fragments come from a zero-padded 16xD f16 LDS image (built once per
// block) -> no per-iteration masking, no B global traffic, low register
// pressure so A-tile loads clause/pipeline.  Single-use weight streams use
// non-temporal loads to keep the 192 MB L2 for reused activations.
// ---------------------------------------------------------------------------

typedef __attribute__((ext_vector_type(16))) _Float16 v16h;
typedef __attribute__((ext_vector_type(8)))  float    v8f;
typedef __attribute__((ext_vector_type(4)))  float    v4f;

__device__ __forceinline__ float sigmoidf_(float v) {
  return 1.0f / (1.0f + __expf(-v));
}

// ===========================================================================
// Tier 1: 64 chained 2-layer LSTMs, hidden 256, 8 steps per frame.
// One persistent workgroup (1024 threads = 32 waves), state in LDS.
// Chain semantics (per reference): next frame h0 = this frame's final CELL,
// c0 = 0.  Weights are reused 8x per frame -> regular-temporal loads.
// ===========================================================================
__global__ __launch_bounds__(1024) void tier1_chain(
    const float* __restrict__ x,
    const float* __restrict__ Wih1, const float* __restrict__ Whh1,
    const float* __restrict__ bih1, const float* __restrict__ bhh1,
    const float* __restrict__ Wih2, const float* __restrict__ Whh2,
    const float* __restrict__ bih2, const float* __restrict__ bhh2,
    float* __restrict__ first_out)
{
  __shared__ float h1[2][256], c1[2][256], h2[2][256], c2[2][256];
  __shared__ float gb[2][1024];
  const int t = threadIdx.x;

  if (t < 512) {
    int b = t >> 8, j = t & 255;
    h1[b][j] = 0.f; c1[b][j] = 0.f; h2[b][j] = 0.f; c2[b][j] = 0.f;
  }
  __syncthreads();

  for (int f = 0; f < 64; ++f) {
    const float* wh1 = Whh1 + (size_t)f * 1024 * 256;
    const float* wi2 = Wih2 + (size_t)f * 1024 * 256;
    const float* wh2 = Whh2 + (size_t)f * 1024 * 256;
    if (f + 1 < 64) {  // prefetch next frame's recurrent weights (1 line/row)
      __builtin_prefetch(Whh1 + (size_t)(f + 1) * 1024 * 256 + (size_t)t * 256, 0, 1);
      __builtin_prefetch(Wih2 + (size_t)(f + 1) * 1024 * 256 + (size_t)t * 256, 0, 1);
      __builtin_prefetch(Whh2 + (size_t)(f + 1) * 1024 * 256 + (size_t)t * 256, 0, 1);
    }
    for (int s = 0; s < 8; ++s) {
      const float x0 = x[(f * 8 + s) * 2 + 0];
      const float x1 = x[(f * 8 + s) * 2 + 1];
      // ---- layer 1 gates: thread t == gate index o (0..1023) ----
      {
        const int o = t;
        const float bsum = bih1[f * 1024 + o] + bhh1[f * 1024 + o];
        const float wih  = Wih1[f * 1024 + o];
        float a0 = bsum + x0 * wih;
        float a1 = bsum + x1 * wih;
        const float4* wr = (const float4*)(wh1 + (size_t)o * 256);
        #pragma unroll 4
        for (int j4 = 0; j4 < 64; ++j4) {
          const float4 w = wr[j4];
          const int j = j4 * 4;
          a0 += w.x * h1[0][j] + w.y * h1[0][j + 1] + w.z * h1[0][j + 2] + w.w * h1[0][j + 3];
          a1 += w.x * h1[1][j] + w.y * h1[1][j + 1] + w.z * h1[1][j + 2] + w.w * h1[1][j + 3];
        }
        gb[0][o] = a0; gb[1][o] = a1;
      }
      __syncthreads();
      if (t < 512) {  // gate order i,f,g,o at offsets 0,256,512,768
        const int b = t >> 8, j = t & 255;
        const float ig = gb[b][j], fg = gb[b][256 + j], gg = gb[b][512 + j], og = gb[b][768 + j];
        const float cn = sigmoidf_(fg) * c1[b][j] + sigmoidf_(ig) * tanhf(gg);
        c1[b][j] = cn;
        h1[b][j] = sigmoidf_(og) * tanhf(cn);
      }
      __syncthreads();
      // ---- layer 2 gates ----
      {
        const int o = t;
        float a0 = bih2[f * 1024 + o] + bhh2[f * 1024 + o];
        float a1 = a0;
        const float4* wi = (const float4*)(wi2 + (size_t)o * 256);
        const float4* wh = (const float4*)(wh2 + (size_t)o * 256);
        #pragma unroll 4
        for (int j4 = 0; j4 < 64; ++j4) {
          const float4 wa = wi[j4];
          const float4 wb = wh[j4];
          const int j = j4 * 4;
          a0 += wa.x * h1[0][j] + wa.y * h1[0][j + 1] + wa.z * h1[0][j + 2] + wa.w * h1[0][j + 3]
              + wb.x * h2[0][j] + wb.y * h2[0][j + 1] + wb.z * h2[0][j + 2] + wb.w * h2[0][j + 3];
          a1 += wa.x * h1[1][j] + wa.y * h1[1][j + 1] + wa.z * h1[1][j + 2] + wa.w * h1[1][j + 3]
              + wb.x * h2[1][j] + wb.y * h2[1][j + 1] + wb.z * h2[1][j + 2] + wb.w * h2[1][j + 3];
        }
        gb[0][o] = a0; gb[1][o] = a1;
      }
      __syncthreads();
      if (t < 512) {
        const int b = t >> 8, j = t & 255;
        const float ig = gb[b][j], fg = gb[b][256 + j], gg = gb[b][512 + j], og = gb[b][768 + j];
        const float cn = sigmoidf_(fg) * c2[b][j] + sigmoidf_(ig) * tanhf(gg);
        c2[b][j] = cn;
        h2[b][j] = sigmoidf_(og) * tanhf(cn);
      }
      __syncthreads();
    }
    // frame boundary: emit output, chain cell -> next hidden, zero cells
    if (t < 512) {
      const int b = t >> 8, j = t & 255;
      first_out[((size_t)f * 2 + b) * 256 + j] = h2[b][j];
      h1[b][j] = c1[b][j]; c1[b][j] = 0.f;
      h2[b][j] = c2[b][j]; c2[b][j] = 0.f;
    }
    __syncthreads();
  }
}

// ===========================================================================
// inp2[t][b][o] = (x[2t,b]+x[2t+1,b])*W2in[o] + dot(first_out[t/4][b], W2hid[o])
// ===========================================================================
__global__ __launch_bounds__(256) void mix12(
    const float* __restrict__ x, const float* __restrict__ W2in,
    const float* __restrict__ W2hid, const float* __restrict__ first_out,
    float* __restrict__ inp2)
{
  __shared__ float fo[256];
  const int t = blockIdx.x, b = blockIdx.y, o = threadIdx.x;
  fo[o] = first_out[(((size_t)(t >> 2)) * 2 + b) * 256 + o];
  __syncthreads();
  const float xs = x[(2 * t) * 2 + b] + x[(2 * t + 1) * 2 + b];
  float acc = xs * W2in[o];
  const float4* wr = (const float4*)(W2hid + (size_t)o * 256);
  #pragma unroll 4
  for (int j4 = 0; j4 < 64; ++j4) {
    const float4 w = wr[j4];
    const int j = 4 * j4;
    acc += w.x * fo[j] + w.y * fo[j + 1] + w.z * fo[j + 2] + w.w * fo[j + 3];
  }
  inp2[((size_t)t * 2 + b) * 256 + o] = acc;
}

// ===========================================================================
// A-tile load: 16x32 f32 weight tile -> f16 fragment, non-temporal (weights
// are single-use; keep L2 for reused data).
// A 16x32 layout (contiguous K, matches B): lanes 0-15 carry K 0-15 of row M=lane,
// lanes 16-31 carry K 16-31 of row M=lane-16.
// ===========================================================================
__device__ __forceinline__ v16h load_a_nt(const float* __restrict__ ar)
{
  const v4f w0 = __builtin_nontemporal_load((const v4f*)(ar));
  const v4f w1 = __builtin_nontemporal_load((const v4f*)(ar + 4));
  const v4f w2 = __builtin_nontemporal_load((const v4f*)(ar + 8));
  const v4f w3 = __builtin_nontemporal_load((const v4f*)(ar + 12));
  v16h a;
  a[0]  = (_Float16)w0[0]; a[1]  = (_Float16)w0[1]; a[2]  = (_Float16)w0[2]; a[3]  = (_Float16)w0[3];
  a[4]  = (_Float16)w1[0]; a[5]  = (_Float16)w1[1]; a[6]  = (_Float16)w1[2]; a[7]  = (_Float16)w1[3];
  a[8]  = (_Float16)w2[0]; a[9]  = (_Float16)w2[1]; a[10] = (_Float16)w2[2]; a[11] = (_Float16)w2[3];
  a[12] = (_Float16)w3[0]; a[13] = (_Float16)w3[1]; a[14] = (_Float16)w3[2]; a[15] = (_Float16)w3[3];
  return a;
}

// ===========================================================================
// Batched GEMV-as-WMMA:  out[i][b][o] = act[i][b][:]*W[i][o][:] + bias(+bias2)
// One wave per 16-output tile; dual accumulators break the WMMA RAW chain.
// B fragments come from a zero-padded [16][D] f16 LDS image: rows 0-1 are the
// real batch rows, rows 2-15 are zeros -> single aligned 32B LDS read per
// fragment, no masking, no divergence.
// ===========================================================================
template<int D, int O, bool RELU, bool HASB2>
__global__ __launch_bounds__(128) void batch2_gemm_wmma(
    const float* __restrict__ act,    // [nmat][2][D]
    const float* __restrict__ W,      // [nmat][O][D]
    const float* __restrict__ bias,   // [nmat][O]
    const float* __restrict__ bias2,  // [nmat][O] or unused
    float* __restrict__ out)          // [nmat][2][O]
{
  constexpr int BPM = O / 64;  // blocks per matrix (4 waves x 16 outputs)
  __shared__ _Float16 bsh[16][D];

  const int tid  = threadIdx.x;
  const int wave = tid >> 5;
  const int lane = tid & 31;
  const int i  = blockIdx.x / BPM;
  const int o0 = ((blockIdx.x % BPM) * 4 + wave) * 16;

  const float* Wm = W   + (size_t)i * O * D;
  const float* am = act + (size_t)i * 2 * D;

  // Build the padded B image: rows 0-1 = f16(act), rows 2-15 = 0.
  {
    unsigned* z = (unsigned*)(&bsh[2][0]);           // 14*D halves = 7*D dwords
    for (int k = tid; k < 7 * D; k += 128) z[k] = 0u;
    _Float16* r = &bsh[0][0];                        // rows 0,1 contiguous = am[0..2D)
    for (int k = tid; k < 2 * D; k += 128) r[k] = (_Float16)am[k];
  }
  __syncthreads();

  const int arow = o0 + (lane & 15);
  const int acol = (lane >> 4) << 4;
  // B layout: lanes 0-15 hold K=0..15 of column N=lane; lanes 16-31 hold K=16..31.
  const int bn   = lane & 15;
  const int bcol = (lane >> 4) << 4;

  const float*    Wrow = Wm + (size_t)arow * D + acol;
  const _Float16* Brow = &bsh[bn][bcol];

  v8f c0 = {}, c1 = {};
  for (int d0 = 0; d0 < D; d0 += 64) {   // D % 64 == 0 at every call site
    const v16h a0 = load_a_nt(Wrow + d0);
    const v16h b0 = *(const v16h*)(Brow + d0);
    c0 = __builtin_amdgcn_wmma_f32_16x16x32_f16(false, a0, false, b0,
                                                (short)0, c0, false, false);
    const v16h a1 = load_a_nt(Wrow + d0 + 32);
    const v16h b1 = *(const v16h*)(Brow + d0 + 32);
    c1 = __builtin_amdgcn_wmma_f32_16x16x32_f16(false, a1, false, b1,
                                                (short)0, c1, false, false);
  }

  // C/D layout: lane l (l<16): D[M=r][N=l] in c[r]; lanes >=16: M = r+8.
  const int mhi = (lane >> 4) << 3;
  if (bn < 2) {
    #pragma unroll
    for (int r = 0; r < 8; ++r) {
      const int o = o0 + mhi + r;
      float v = c0[r] + c1[r] + bias[(size_t)i * O + o];
      if (HASB2) v += bias2[(size_t)i * O + o];
      if (RELU)  v = fmaxf(v, 0.0f);
      out[((size_t)i * 2 + bn) * O + o] = v;
    }
  }
}

// ===========================================================================
// Tier 2: 256 chained 2-layer LSTMs (hidden 128), one step each.  Input-gate
// contribution (incl. both biases) is precomputed by the WMMA kernel above.
// Per-step weights are single-use -> non-temporal loads.
// ===========================================================================
__global__ __launch_bounds__(512) void tier2_chain(
    const float* __restrict__ gates_in,  // [256][2][512]  (Wih1 part + biases)
    const float* __restrict__ Whh1, const float* __restrict__ Wih2,
    const float* __restrict__ Whh2,
    const float* __restrict__ bih2, const float* __restrict__ bhh2,
    float* __restrict__ second_out)      // [256][2][128]
{
  __shared__ float h1[2][128], c1[2][128], h2[2][128], c2[2][128];
  __shared__ float gb[2][512];
  const int t = threadIdx.x;
  if (t < 256) {
    const int b = t >> 7, j = t & 127;
    h1[b][j] = 0.f; c1[b][j] = 0.f; h2[b][j] = 0.f; c2[b][j] = 0.f;
  }
  __syncthreads();

  for (int step = 0; step < 256; ++step) {
    const float* wh1 = Whh1 + (size_t)step * 512 * 128;
    const float* wi2 = Wih2 + (size_t)step * 512 * 128;
    const float* wh2 = Whh2 + (size_t)step * 512 * 128;
    if (step + 1 < 256) {
      __builtin_prefetch(Whh1 + (size_t)(step + 1) * 512 * 128 + (size_t)t * 128, 0, 1);
      __builtin_prefetch(Wih2 + (size_t)(step + 1) * 512 * 128 + (size_t)t * 128, 0, 1);
      __builtin_prefetch(Whh2 + (size_t)(step + 1) * 512 * 128 + (size_t)t * 128, 0, 1);
    }
    // ---- layer 1: recurrent part only ----
    {
      const int o = t;
      float a0 = gates_in[((size_t)step * 2 + 0) * 512 + o];
      float a1 = gates_in[((size_t)step * 2 + 1) * 512 + o];
      const v4f* wr = (const v4f*)(wh1 + (size_t)o * 128);
      #pragma unroll 4
      for (int j4 = 0; j4 < 32; ++j4) {
        const v4f w = __builtin_nontemporal_load(wr + j4);
        const int j = 4 * j4;
        a0 += w[0] * h1[0][j] + w[1] * h1[0][j + 1] + w[2] * h1[0][j + 2] + w[3] * h1[0][j + 3];
        a1 += w[0] * h1[1][j] + w[1] * h1[1][j + 1] + w[2] * h1[1][j + 2] + w[3] * h1[1][j + 3];
      }
      gb[0][o] = a0; gb[1][o] = a1;
    }
    __syncthreads();
    if (t < 256) {  // gate order i,f,g,o at offsets 0,128,256,384
      const int b = t >> 7, j = t & 127;
      const float ig = gb[b][j], fg = gb[b][128 + j], gg = gb[b][256 + j], og = gb[b][384 + j];
      const float cn = sigmoidf_(fg) * c1[b][j] + sigmoidf_(ig) * tanhf(gg);
      c1[b][j] = cn;
      h1[b][j] = sigmoidf_(og) * tanhf(cn);
    }
    __syncthreads();
    // ---- layer 2 ----
    {
      const int o = t;
      const float base = bih2[step * 512 + o] + bhh2[step * 512 + o];
      float a0 = base, a1 = base;
      const v4f* wi = (const v4f*)(wi2 + (size_t)o * 128);
      const v4f* wh = (const v4f*)(wh2 + (size_t)o * 128);
      #pragma unroll 4
      for (int j4 = 0; j4 < 32; ++j4) {
        const v4f wa = __builtin_nontemporal_load(wi + j4);
        const v4f wb = __builtin_nontemporal_load(wh + j4);
        const int j = 4 * j4;
        a0 += wa[0] * h1[0][j] + wa[1] * h1[0][j + 1] + wa[2] * h1[0][j + 2] + wa[3] * h1[0][j + 3]
            + wb[0] * h2[0][j] + wb[1] * h2[0][j + 1] + wb[2] * h2[0][j + 2] + wb[3] * h2[0][j + 3];
        a1 += wa[0] * h1[1][j] + wa[1] * h1[1][j + 1] + wa[2] * h1[1][j + 2] + wa[3] * h1[1][j + 3]
            + wb[0] * h2[1][j] + wb[1] * h2[1][j + 1] + wb[2] * h2[1][j + 2] + wb[3] * h2[1][j + 3];
      }
      gb[0][o] = a0; gb[1][o] = a1;
    }
    __syncthreads();
    if (t < 256) {  // update layer2, emit, and chain (cell -> next hidden)
      const int b = t >> 7, j = t & 127;
      const float ig = gb[b][j], fg = gb[b][128 + j], gg = gb[b][256 + j], og = gb[b][384 + j];
      const float cn = sigmoidf_(fg) * c2[b][j] + sigmoidf_(ig) * tanhf(gg);
      const float hn = sigmoidf_(og) * tanhf(cn);
      second_out[((size_t)step * 2 + b) * 128 + j] = hn;
      h2[b][j] = cn; c2[b][j] = 0.f;        // next-step h0 = this cell
      h1[b][j] = c1[b][j]; c1[b][j] = 0.f;
    }
    __syncthreads();
  }
}

// ===========================================================================
// inp3[s][b][o] = x[s,b]*W3in[o] + dot(second_out[s/4][b], W3hid[o])
// ===========================================================================
__global__ __launch_bounds__(512) void mix23(
    const float* __restrict__ x, const float* __restrict__ W3in,
    const float* __restrict__ W3hid, const float* __restrict__ second_out,
    float* __restrict__ inp3)
{
  __shared__ float so[128];
  const int s = blockIdx.x, b = blockIdx.y, o = threadIdx.x;
  if (o < 128) so[o] = second_out[(((size_t)(s >> 2)) * 2 + b) * 128 + o];
  __syncthreads();
  float acc = x[s * 2 + b] * W3in[o];
  const float4* wr = (const float4*)(W3hid + (size_t)o * 128);
  #pragma unroll 4
  for (int j4 = 0; j4 < 32; ++j4) {
    const float4 w = wr[j4];
    const int j = 4 * j4;
    acc += w.x * so[j] + w.y * so[j + 1] + w.z * so[j + 2] + w.w * so[j + 3];
  }
  inp3[((size_t)s * 2 + b) * 512 + o] = acc;
}

// ===========================================================================
// Tier-3 output layer + softmax over batch axis (keep b=1):
// y[i][b] = relu(dot(h2[i][b], W3[i]) + b3[i]); out[i] = e^y1/(e^y0+e^y1)
// ===========================================================================
__global__ __launch_bounds__(64) void tier3_out(
    const float* __restrict__ h2buf, const float* __restrict__ W3,
    const float* __restrict__ b3, float* __restrict__ out)
{
  __shared__ float part[64];
  const int i = blockIdx.x;
  const int t = threadIdx.x;
  const int b = t >> 5, lane = t & 31;
  const float* act = h2buf + ((size_t)i * 2 + b) * 1024;
  const float* w   = W3 + (size_t)i * 1024;
  float acc = 0.f;
  for (int k = lane; k < 1024; k += 32) acc += act[k] * w[k];
  part[t] = acc;
  __syncthreads();
  if (t == 0) {
    float y0 = 0.f, y1 = 0.f;
    #pragma unroll
    for (int k = 0; k < 32; ++k) { y0 += part[k]; y1 += part[32 + k]; }
    y0 = fmaxf(y0 + b3[i], 0.f);
    y1 = fmaxf(y1 + b3[i], 0.f);
    const float e0 = __expf(y0), e1 = __expf(y1);
    out[i] = e1 / (e0 + e1);
  }
}

// ===========================================================================
extern "C" void kernel_launch(void* const* d_in, const int* in_sizes, int n_in,
                              void* d_out, int out_size, void* d_ws, size_t ws_size,
                              hipStream_t stream) {
  (void)in_sizes; (void)n_in; (void)out_size; (void)ws_size;
  const float* x        = (const float*)d_in[0];
  const float* t1_Wih1  = (const float*)d_in[1];
  const float* t1_Whh1  = (const float*)d_in[2];
  const float* t1_bih1  = (const float*)d_in[3];
  const float* t1_bhh1  = (const float*)d_in[4];
  const float* t1_Wih2  = (const float*)d_in[5];
  const float* t1_Whh2  = (const float*)d_in[6];
  const float* t1_bih2  = (const float*)d_in[7];
  const float* t1_bhh2  = (const float*)d_in[8];
  const float* W2in     = (const float*)d_in[9];
  const float* W2hid    = (const float*)d_in[10];
  const float* t2_Wih1  = (const float*)d_in[11];
  const float* t2_Whh1  = (const float*)d_in[12];
  const float* t2_bih1  = (const float*)d_in[13];
  const float* t2_bhh1  = (const float*)d_in[14];
  const float* t2_Wih2  = (const float*)d_in[15];
  const float* t2_Whh2  = (const float*)d_in[16];
  const float* t2_bih2  = (const float*)d_in[17];
  const float* t2_bhh2  = (const float*)d_in[18];
  const float* W3in     = (const float*)d_in[19];
  const float* W3hid    = (const float*)d_in[20];
  const float* t3_W1    = (const float*)d_in[21];
  const float* t3_b1    = (const float*)d_in[22];
  const float* t3_W2    = (const float*)d_in[23];
  const float* t3_b2    = (const float*)d_in[24];
  const float* t3_W3    = (const float*)d_in[25];
  const float* t3_b3    = (const float*)d_in[26];

  float* ws         = (float*)d_ws;
  float* first_out  = ws;                      // 64*2*256    = 32768
  float* inp2       = first_out + 32768;       // 256*2*256   = 131072
  float* gates2     = inp2 + 131072;           // 256*2*512   = 262144
  float* second_out = gates2 + 262144;         // 256*2*128   = 65536
  float* inp3       = second_out + 65536;      // 512*2*512   = 524288
  float* h1buf      = inp3 + 524288;           // 512*2*1024  = 1048576
  float* h2buf      = h1buf + 1048576;         // 512*2*1024  = 1048576

  // Tier 1 chain (latency-bound, 1 persistent workgroup)
  tier1_chain<<<1, 1024, 0, stream>>>(x, t1_Wih1, t1_Whh1, t1_bih1, t1_bhh1,
                                      t1_Wih2, t1_Whh2, t1_bih2, t1_bhh2, first_out);
  // Tier1 -> Tier2 mixing
  mix12<<<dim3(256, 2), 256, 0, stream>>>(x, W2in, W2hid, first_out, inp2);
  // Tier2 input gates, hoisted out of the chain (WMMA, 134 MB stream)
  batch2_gemm_wmma<256, 512, false, true>
      <<<256 * 8, 128, 0, stream>>>(inp2, t2_Wih1, t2_bih1, t2_bhh1, gates2);
  // Tier 2 chain (latency-bound)
  tier2_chain<<<1, 512, 0, stream>>>(gates2, t2_Whh1, t2_Wih2, t2_Whh2,
                                     t2_bih2, t2_bhh2, second_out);
  // Tier2 -> Tier3 mixing
  mix23<<<dim3(512, 2), 512, 0, stream>>>(x, W3in, W3hid, second_out, inp3);
  // Tier 3 MLPs (WMMA, 3.2 GB stream — the bandwidth bulk)
  batch2_gemm_wmma<512, 1024, true, false>
      <<<512 * 16, 128, 0, stream>>>(inp3, t3_W1, t3_b1, nullptr, h1buf);
  batch2_gemm_wmma<1024, 1024, true, false>
      <<<512 * 16, 128, 0, stream>>>(h1buf, t3_W2, t3_b2, nullptr, h2buf);
  // Output layer + softmax over the 2-element batch axis
  tier3_out<<<512, 64, 0, stream>>>(h2buf, t3_W3, t3_b3, (float*)d_out);
}
